// HMLSTM_26293789786950
// MI455X (gfx1250) — matrix operand order repaired and checked by reading, not compiled, not run on hardware
//
#include <hip/hip_runtime.h>

#define D_ 512
#define B_ 64
#define T_ 512
#define ROWS_ (4 * D_ + 1)  // 2049

typedef __bf16 bf16x16 __attribute__((ext_vector_type(16)));
typedef float f32x8 __attribute__((ext_vector_type(8)));
typedef unsigned int uint4v __attribute__((ext_vector_type(4)));

union FragU { bf16x16 bf; uint4v u[2]; };

__device__ __forceinline__ unsigned short f2bf(float f) {
  unsigned int u = __float_as_uint(f);
  u += 0x7FFFu + ((u >> 16) & 1u);  // round-to-nearest-even
  return (unsigned short)(u >> 16);
}
__device__ __forceinline__ float sigmoidf_(float x) {
  return 1.0f / (1.0f + __expf(-x));
}

// A-operand fragment (16x32 bf16): per-lane K = {c0..c0+7, c0+16..c0+23}
__device__ __forceinline__ bf16x16 ldfragA(const unsigned short* p) {
  FragU f;
  f.u[0] = *(const uint4v*)(p);
  f.u[1] = *(const uint4v*)(p + 16);
  return f.bf;
}
// B-operand fragment (32x16 bf16): per-lane K = {c0..c0+7, c0+16..c0+23}
__device__ __forceinline__ bf16x16 ldfragB(const unsigned short* p) {
  FragU f;
  f.u[0] = *(const uint4v*)(p);
  f.u[1] = *(const uint4v*)(p + 16);
  return f.bf;
}

// 8-wide bf16 dot for the scalar z-row path
__device__ __forceinline__ float dot8(uint4v a, uint4v b) {
  float s = 0.f;
#pragma unroll
  for (int j = 0; j < 4; ++j) {
    s += __uint_as_float(a[j] << 16) * __uint_as_float(b[j] << 16);
    s += __uint_as_float(a[j] & 0xFFFF0000u) * __uint_as_float(b[j] & 0xFFFF0000u);
  }
  return s;
}

__global__ void cvt_bf16_kernel(const float* __restrict__ s,
                                unsigned short* __restrict__ d, int n) {
  int i = blockIdx.x * blockDim.x + threadIdx.x;
  if (i < n) d[i] = f2bf(s[i]);
}

// Sense-reversing grid barrier (33 workgroups, all co-resident).
__device__ __forceinline__ void grid_sync(unsigned* count, unsigned* sense,
                                          unsigned nb) {
  __threadfence();  // release this thread's prior writes to device scope
  __syncthreads();
  if (threadIdx.x == 0) {
    unsigned s = __hip_atomic_load(sense, __ATOMIC_RELAXED, __HIP_MEMORY_SCOPE_AGENT);
    if (__hip_atomic_fetch_add(count, 1u, __ATOMIC_ACQ_REL, __HIP_MEMORY_SCOPE_AGENT) ==
        nb - 1u) {
      __hip_atomic_store(count, 0u, __ATOMIC_RELAXED, __HIP_MEMORY_SCOPE_AGENT);
      __hip_atomic_fetch_add(sense, 1u, __ATOMIC_RELEASE, __HIP_MEMORY_SCOPE_AGENT);
    } else {
      while (__hip_atomic_load(sense, __ATOMIC_ACQUIRE, __HIP_MEMORY_SCOPE_AGENT) == s) {
        __builtin_amdgcn_s_sleep(1);
      }
    }
    __threadfence();  // acquire: refresh near caches for the WGP
  }
  __syncthreads();
}

// Grid: 33 blocks x 128 threads. Blocks 0..31: 4 tile-waves each (one per
// SIMD32 of a WGP -> each wave owns a WMMA pipe). Block 32: z-row.
__global__ __launch_bounds__(128) void hmlstm_persistent_kernel(
    const unsigned short* __restrict__ wb,   // [3][2049][512] bf16
    const unsigned short* __restrict__ ub,   // [3][2049][512]
    const unsigned short* __restrict__ utb,  // [2][2049][512]
    const float* __restrict__ b0, const float* __restrict__ b1,
    const float* __restrict__ b2,
    const unsigned short* __restrict__ xb,  // (B,T,D) bf16
    unsigned short* __restrict__ hT,        // [2][3][64][512] bf16
    float* __restrict__ hF,                 // [2][3][64][512] fp32
    float* __restrict__ cS,                 // [3][64][512] fp32
    float* __restrict__ zS,                 // [2][3][64] fp32
    float* __restrict__ out,                // h_all | c_all | z_all
    unsigned* __restrict__ bar)             // [2]: count, sense
{
  __shared__ float red[128];
  const int lane = threadIdx.x & 31;
  const int lh = lane & 15;
  const int hi = lane >> 4;  // 0 or 1
  const bool zblock = (blockIdx.x == 32);

  // Tile coords (blocks 0..31): mt = batch tile 0..3, nt = feature tile 0..31
  const int tile = blockIdx.x * 4 + (threadIdx.x >> 5);
  const int mt = (tile >> 5) & 3;
  const int nt = tile & 31;
  const int d = nt * 16 + lh;
  const int woff = (nt * 16 + lh) * 512 + hi * 8;

  const int WSZ = ROWS_ * D_;  // 1,049,088
  const int HSZ = B_ * D_;     // 32,768
  const size_t BTD = (size_t)B_ * T_ * D_;

  const f32x8 z8 = {0.f, 0.f, 0.f, 0.f, 0.f, 0.f, 0.f, 0.f};

  for (int t = 0; t < T_; ++t) {
    const int pp = t & 1, cc = pp ^ 1;
#pragma unroll
    for (int l = 0; l < 3; ++l) {
      const bool HAS_TOP = (l < 2);
      const unsigned short* Wg = wb + l * WSZ;
      const unsigned short* Ug = ub + l * WSZ;
      const unsigned short* Utg = HAS_TOP ? (utb + l * WSZ) : nullptr;
      const float* bias = (l == 0) ? b0 : ((l == 1) ? b1 : b2);
      const unsigned short* hbot;
      int hbs;
      if (l == 0) { hbot = xb + t * D_; hbs = T_ * D_; }
      else        { hbot = hT + (cc * 3 + (l - 1)) * HSZ; hbs = D_; }
      const unsigned short* hprev = hT + (pp * 3 + l) * HSZ;
      const unsigned short* htop = HAS_TOP ? (hT + (pp * 3 + (l + 1)) * HSZ) : nullptr;
      const float* zprev = zS + (pp * 3 + l) * B_;
      const float* zbot = (l == 0) ? nullptr : (zS + (cc * 3 + (l - 1)) * B_);
      float* cstate = cS + l * HSZ;
      const float* hprevF = hF + (pp * 3 + l) * HSZ;
      float* hcurF = hF + (cc * 3 + l) * HSZ;
      unsigned short* hcurT = hT + (cc * 3 + l) * HSZ;
      float* zcur = zS + (cc * 3 + l) * B_;
      float* h_out = out + (size_t)l * BTD;
      float* c_out = out + 3 * BTD + (size_t)l * BTD;
      float* z_out = out + 6 * BTD + (size_t)l * B_ * T_;

      if (zblock) {
        // --- z-row (row 2048), split-K over 2 chunks of 256, vectorized ---
        const int b = threadIdx.x & 63;
        const int ck = threadIdx.x >> 6;  // 0 or 1
        const unsigned short* wr = Wg + 2048 * 512 + ck * 256;
        const unsigned short* ur = Ug + 2048 * 512 + ck * 256;
        const unsigned short* hb = hbot + b * hbs + ck * 256;
        const unsigned short* hp = hprev + b * 512 + ck * 256;
        float aw = 0.f, au = 0.f;
#pragma unroll 4
        for (int k = 0; k < 256; k += 8) {
          aw += dot8(*(const uint4v*)(wr + k), *(const uint4v*)(hb + k));
          au += dot8(*(const uint4v*)(ur + k), *(const uint4v*)(hp + k));
        }
        if (HAS_TOP) {
          const unsigned short* utr = Utg + 2048 * 512 + ck * 256;
          const unsigned short* ht = htop + b * 512 + ck * 256;
#pragma unroll 4
          for (int k = 0; k < 256; k += 8)
            au += dot8(*(const uint4v*)(utr + k), *(const uint4v*)(ht + k));
        }
        red[threadIdx.x] = aw + zprev[b] * au;
        __syncthreads();
        if (ck == 0) {
          float s = red[b] + red[b + 64] + bias[2048];
          float zhat = fminf(fmaxf((s + 1.0f) * 0.5f, 0.0f), 1.0f);
          float hard = (zhat > 0.5f) ? 1.0f : 0.0f;  // STE forward value
          zcur[b] = hard;
          z_out[b * T_ + t] = hard;
        }
      } else {
        // --- WMMA tile: 16 batches x 16 features x 4 gates ---
        const unsigned short* aB = hbot + (mt * 16 + lh) * hbs + hi * 8;
        const unsigned short* aH = hprev + (mt * 16 + lh) * 512 + hi * 8;
        const unsigned short* aT =
            HAS_TOP ? (htop + (mt * 16 + lh) * 512 + hi * 8) : aH;

        f32x8 acc0[4] = {z8, z8, z8, z8};  // W @ h_bottom (transposed tile)
        f32x8 acc1[4] = {z8, z8, z8, z8};  // U @ h + Ut @ h_top

        for (int k0 = 0; k0 < 512; k0 += 32) {
          bf16x16 fB = ldfragA(aB + k0);
          bf16x16 fH = ldfragA(aH + k0);
          bf16x16 fT = fH;
          if (HAS_TOP) fT = ldfragA(aT + k0);
#pragma unroll
          for (int gp = 0; gp < 2; ++gp) {
            const int g0 = gp * 2, g1 = g0 + 1;
            const int o0 = g0 * (512 * 512) + woff + k0;
            const int o1 = g1 * (512 * 512) + woff + k0;
            // Batch all fragment loads for the gate pair, then WMMA them:
            bf16x16 w0 = ldfragB(Wg + o0);
            bf16x16 w1 = ldfragB(Wg + o1);
            bf16x16 u0 = ldfragB(Ug + o0);
            bf16x16 u1 = ldfragB(Ug + o1);
            bf16x16 t0 = w0, t1 = w1;
            if (HAS_TOP) { t0 = ldfragB(Utg + o0); t1 = ldfragB(Utg + o1); }
            acc0[g0] = __builtin_amdgcn_wmma_f32_16x16x32_bf16(
                false, fB, false, w0, (short)0, acc0[g0], false, false);
            acc0[g1] = __builtin_amdgcn_wmma_f32_16x16x32_bf16(
                false, fB, false, w1, (short)0, acc0[g1], false, false);
            acc1[g0] = __builtin_amdgcn_wmma_f32_16x16x32_bf16(
                false, fH, false, u0, (short)0, acc1[g0], false, false);
            acc1[g1] = __builtin_amdgcn_wmma_f32_16x16x32_bf16(
                false, fH, false, u1, (short)0, acc1[g1], false, false);
            if (HAS_TOP) {
              acc1[g0] = __builtin_amdgcn_wmma_f32_16x16x32_bf16(
                  false, fT, false, t0, (short)0, acc1[g0], false, false);
              acc1[g1] = __builtin_amdgcn_wmma_f32_16x16x32_bf16(
                  false, fT, false, t1, (short)0, acc1[g1], false, false);
            }
          }
        }

        // Epilogue: elem i, lane l -> b = mt*16 + (hi ? i+8 : i), feature d
        const float bf_ = bias[0 * 512 + d];
        const float bi_ = bias[1 * 512 + d];
        const float bo_ = bias[2 * 512 + d];
        const float bg_ = bias[3 * 512 + d];
#pragma unroll
        for (int i = 0; i < 8; ++i) {
          const int b = mt * 16 + (hi ? (i + 8) : i);
          const float z = zprev[b];
          const float zb = zbot ? zbot[b] : 1.0f;
          const float f = sigmoidf_(acc0[0][i] + z * acc1[0][i] + bf_);
          const float ii = sigmoidf_(acc0[1][i] + z * acc1[1][i] + bi_);
          const float o = sigmoidf_(acc0[2][i] + z * acc1[2][i] + bo_);
          const float g = tanhf(acc0[3][i] + z * acc1[3][i] + bg_);
          const int idx = b * 512 + d;
          const float cp = cstate[idx];
          const float ig = ii * g;
          const float omz = 1.0f - z;
          const float cn = z * ig + omz * (1.0f - zb) * cp + omz * zb * (f * cp + ig);
          const float tc = tanhf(cn);
          const float hp = hprevF[idx];
          const float otc = o * tc;
          const float hn = z * otc + omz * (1.0f - zb) * hp + omz * zb * otc;
          cstate[idx] = cn;
          hcurF[idx] = hn;
          hcurT[idx] = f2bf(hn);
          const int oidx = (b * T_ + t) * D_ + d;
          h_out[oidx] = hn;
          c_out[oidx] = cn;
        }
      }
      grid_sync(bar, bar + 1, 33u);
    }
  }
}

extern "C" void kernel_launch(void* const* d_in, const int* in_sizes, int n_in,
                              void* d_out, int out_size, void* d_ws,
                              size_t ws_size, hipStream_t stream) {
  const float* x = (const float*)d_in[0];
  const float* W[3] = {(const float*)d_in[1], (const float*)d_in[2], (const float*)d_in[3]};
  const float* U[3] = {(const float*)d_in[4], (const float*)d_in[5], (const float*)d_in[6]};
  const float* Ut[2] = {(const float*)d_in[7], (const float*)d_in[8]};
  const float* b0 = (const float*)d_in[9];
  const float* b1 = (const float*)d_in[10];
  const float* b2 = (const float*)d_in[11];

  const size_t WSZ = (size_t)ROWS_ * D_;    // 1,049,088 elems per weight matrix
  const size_t XSZ = (size_t)B_ * T_ * D_;  // 16,777,216
  const size_t HSZ = (size_t)B_ * D_;       // 32,768

  char* p = (char*)d_ws;
  auto carve = [&](size_t bytes) -> char* {
    char* r = p;
    p += (bytes + 255) & ~(size_t)255;
    return r;
  };
  unsigned short* wb = (unsigned short*)carve(3 * WSZ * 2);
  unsigned short* ub = (unsigned short*)carve(3 * WSZ * 2);
  unsigned short* utb = (unsigned short*)carve(2 * WSZ * 2);
  unsigned short* xb = (unsigned short*)carve(XSZ * 2);
  unsigned short* hT = (unsigned short*)carve(2 * 3 * HSZ * 2);
  float* hF = (float*)carve(2 * 3 * HSZ * 4);
  float* cS = (float*)carve(3 * HSZ * 4);
  float* zS = (float*)carve(2 * 3 * B_ * 4);
  unsigned* bar = (unsigned*)carve(2 * sizeof(unsigned));

  auto cvt = [&](const float* s, unsigned short* d, size_t n) {
    cvt_bf16_kernel<<<(int)((n + 255) / 256), 256, 0, stream>>>(s, d, (int)n);
  };
  for (int l = 0; l < 3; ++l) cvt(W[l], wb + l * WSZ, WSZ);
  for (int l = 0; l < 3; ++l) cvt(U[l], ub + l * WSZ, WSZ);
  for (int l = 0; l < 2; ++l) cvt(Ut[l], utb + l * WSZ, WSZ);
  cvt(x, xb, XSZ);

  hipMemsetAsync(hT, 0, 2 * 3 * HSZ * 2, stream);
  hipMemsetAsync(hF, 0, 2 * 3 * HSZ * 4, stream);
  hipMemsetAsync(cS, 0, 3 * HSZ * 4, stream);
  hipMemsetAsync(zS, 0, 2 * 3 * B_ * 4, stream);
  hipMemsetAsync(bar, 0, 2 * sizeof(unsigned), stream);

  hmlstm_persistent_kernel<<<33, 128, 0, stream>>>(
      wb, ub, utb, b0, b1, b2, xb, hT, hF, cS, zS, (float*)d_out, bar);
}
